// SwinTransformerBlock3D_64098091926064
// MI455X (gfx1250) — matrix operand order repaired and checked by reading, not compiled, not run on hardware
//
#include <hip/hip_runtime.h>
#include <hip/hip_bf16.h>
#include <math.h>

typedef __bf16 bf16;
typedef __attribute__((ext_vector_type(16))) __bf16 bf16x16;
typedef __attribute__((ext_vector_type(8)))  float  f32x8;
typedef __attribute__((ext_vector_type(4)))  float  f32x4;
typedef __attribute__((ext_vector_type(4)))  unsigned u32x4;
typedef int v4i_ __attribute__((vector_size(16)));   // matches async-LDS builtin pointee

#define DEVINL __device__ __forceinline__

// -------- CDNA5 async global->LDS copy (guarded; falls back to b128 copy) -----
#if __has_builtin(__builtin_amdgcn_global_load_async_to_lds_b128) && \
    __has_builtin(__builtin_amdgcn_s_wait_asynccnt)
#define USE_ASYNC_LDS 1
#else
#define USE_ASYNC_LDS 0
#endif

DEVINL void cp_b128_to_lds(const void* g, void* l) {
#if USE_ASYNC_LDS
  __builtin_amdgcn_global_load_async_to_lds_b128(
      (__attribute__((address_space(1))) v4i_*)(v4i_*)(void*)g,
      (__attribute__((address_space(3))) v4i_*)(v4i_*)l, 0, 0);
#else
  *(u32x4*)l = *(const u32x4*)g;
#endif
}
DEVINL void cp_lds_wait() {
#if USE_ASYNC_LDS
  __builtin_amdgcn_s_wait_asynccnt(0);
#endif
}

// ---------------- problem constants ----------------
constexpr int B_ = 2, C_ = 96, D_ = 32, H_ = 64, W_ = 64;
constexpr int HEADS = 4, HD = 24, WSZ = 4, SHIFT = 2, N_TOK = 64;
constexpr int DHW = D_ * H_ * W_;                   // 131072
constexpr int NWD = D_ / WSZ, NWH = H_ / WSZ, NWW = W_ / WSZ; // 8,16,16
constexpr int NW = NWD * NWH * NWW;                 // 2048 windows / batch
constexpr int ROWS = B_ * NW * N_TOK;               // 262144 tokens
constexpr float SCALE = 0.20412414523193154f;       // 24^-0.5
constexpr int RPB_N = 343 * HEADS;                  // rel-pos bias table elems

// ---------------- workspace layout (bytes) ----------------
constexpr size_t OFF_QKVW = 0;
constexpr size_t OFF_PROJW = OFF_QKVW + (size_t)3 * C_ * C_ * 2;
constexpr size_t OFF_FC1W  = OFF_PROJW + (size_t)C_ * C_ * 2;
constexpr size_t OFF_FC2W  = OFF_FC1W + (size_t)4 * C_ * C_ * 2;
constexpr size_t OFF_XW    = OFF_FC2W + (size_t)4 * C_ * C_ * 2;
constexpr size_t OFF_QKV   = OFF_XW + (size_t)ROWS * C_ * 2;
constexpr size_t OFF_AO    = OFF_QKV + (size_t)ROWS * 3 * C_ * 2;
constexpr size_t OFF_Y     = OFF_AO + (size_t)ROWS * C_ * 2;

// ---------------- WMMA fragment helpers (wave32, contiguous k = 16*hi + s in A and B) ----------------
union FragB16 { bf16x16 v; unsigned u[8]; bf16 e[16]; };

// A-matrix 16x32 bf16 from row-major LDS tile (stride `ld` elems, even).
DEVINL bf16x16 frag_a_lds(const bf16* p, int ld) {
  int lane = threadIdx.x & 31;
  int m = lane & 15, hi = lane >> 4;
  FragB16 f;
#pragma unroll
  for (int v = 0; v < 8; ++v) {
    int k = 2 * v + hi * 16;
    f.u[v] = *(const unsigned*)(p + m * ld + k);
  }
  return f.v;
}

// B-matrix 32x16 bf16 from (n,k) row-major LDS tile (weight layout).
DEVINL bf16x16 frag_b_lds(const bf16* p, int ld) {
  int lane = threadIdx.x & 31;
  int n = lane & 15, hi = lane >> 4;
  FragB16 f;
#pragma unroll
  for (int v = 0; v < 8; ++v) {
    int k = 2 * v + hi * 16;
    f.u[v] = *(const unsigned*)(p + n * ld + k);
  }
  return f.v;
}

// B-matrix 32x16 bf16 from (k,n) row-major LDS tile (paired halfword gather).
DEVINL bf16x16 frag_b_kxn(const bf16* p, int ld) {
  int lane = threadIdx.x & 31;
  int n = lane & 15, hi = lane >> 4;
  FragB16 f;
#pragma unroll
  for (int w = 0; w < 8; ++w) {
    int k = 2 * w + hi * 16;
    f.e[2 * w]     = p[k * ld + n];
    f.e[2 * w + 1] = p[(k + 1) * ld + n];
  }
  return f.v;
}

DEVINL f32x8 wmma_bf16(bf16x16 a, bf16x16 b, f32x8 c) {
  return __builtin_amdgcn_wmma_f32_16x16x32_bf16(false, a, false, b, (short)0, c,
                                                 false, false);
}

DEVINL f32x8 zero8() { f32x8 z; for (int i = 0; i < 8; ++i) z[i] = 0.f; return z; }

// ---------------- K0: f32 -> bf16 weight conversion ----------------
__global__ void k_convert_weights(const float* __restrict__ qkvw,
                                  const float* __restrict__ projw,
                                  const float* __restrict__ fc1w,
                                  const float* __restrict__ fc2w,
                                  bf16* o_qkv, bf16* o_proj, bf16* o_fc1, bf16* o_fc2) {
  int i = blockIdx.x * blockDim.x + threadIdx.x;
  if (i < 3 * C_ * C_) o_qkv[i] = (bf16)qkvw[i];
  if (i < C_ * C_)     o_proj[i] = (bf16)projw[i];
  if (i < 4 * C_ * C_) { o_fc1[i] = (bf16)fc1w[i]; o_fc2[i] = (bf16)fc2w[i]; }
}

// ---------------- K1: LN1 + roll(-SHIFT) + window partition -> bf16 ----------------
__global__ void k_ln1_window(const float* __restrict__ x,
                             const float* __restrict__ g, const float* __restrict__ bta,
                             bf16* __restrict__ xw) {
  __shared__ float sv[C_ * 128];  // 48KB
  int tid = threadIdx.x;
  int blk = blockIdx.x;
  int b = blk >> 10;                        // 1024 blocks / batch
  int posBase = (blk & 1023) * 128;
  const float* xb = x + (size_t)b * C_ * DHW;
#pragma unroll 4
  for (int c = 0; c < C_; ++c)
    sv[c * 128 + tid] = xb[(size_t)c * DHW + posBase + tid];
  float s = 0.f, s2 = 0.f;
#pragma unroll 4
  for (int c = 0; c < C_; ++c) { float v = sv[c * 128 + tid]; s += v; s2 += v * v; }
  float m = s * (1.0f / C_);
  float var = s2 * (1.0f / C_) - m * m;
  float rs = rsqrtf(var + 1e-5f);
  int pos = posBase + tid;
  int w = pos & 63, h = (pos >> 6) & 63, d = pos >> 12;
  int dr = (d - SHIFT) & (D_ - 1), hr = (h - SHIFT) & (H_ - 1), wr = (w - SHIFT) & (W_ - 1);
  int wd = dr >> 2, wh = hr >> 2, ww = wr >> 2;
  int tok = ((dr & 3) << 4) | ((hr & 3) << 2) | (wr & 3);
  int win = ((b * NWD + wd) * NWH + wh) * NWW + ww;
  bf16 ov[C_];
#pragma unroll
  for (int c = 0; c < C_; ++c)
    ov[c] = (bf16)((sv[c * 128 + tid] - m) * rs * g[c] + bta[c]);
  bf16* dst = xw + ((size_t)win * N_TOK + tok) * C_;   // 192B row, 16B aligned
#pragma unroll
  for (int c8 = 0; c8 < 12; ++c8)
    *(u32x4*)(dst + c8 * 8) = *(const u32x4*)(ov + c8 * 8);
}

// ---------------- K2: QKV GEMM (262144x96)@(96x288), async b128 staging ---------
__global__ __launch_bounds__(256) void k_qkv_gemm(const bf16* __restrict__ xw,
                                                  const bf16* __restrict__ wq,
                                                  const float* __restrict__ bias,
                                                  bf16* __restrict__ qkv) {
  __shared__ __align__(16) bf16 aT[128 * 32];   // 8KB
  __shared__ __align__(16) bf16 bT[96 * 32];    // 6KB
  __shared__ __align__(16) bf16 oT[128 * 96];   // 24KB output transpose tile
  int tid = threadIdx.x;
  int wave = tid >> 5, wm = wave >> 1, wn = wave & 1;
  int m0 = blockIdx.x * 128;
  int n0 = blockIdx.y * 96;
  f32x8 acc[2][3];
  for (int i = 0; i < 2; ++i) for (int j = 0; j < 3; ++j) acc[i][j] = zero8();
  for (int ks = 0; ks < 3; ++ks) {
    for (int i = tid; i < 512; i += 256) {      // 128x32 bf16 in 8-elem chunks
      int r = i >> 2, c8 = (i & 3) * 8;
      cp_b128_to_lds(xw + (size_t)(m0 + r) * C_ + ks * 32 + c8, aT + r * 32 + c8);
    }
    for (int i = tid; i < 384; i += 256) {      // 96x32
      int r = i >> 2, c8 = (i & 3) * 8;
      cp_b128_to_lds(wq + (size_t)(n0 + r) * C_ + ks * 32 + c8, bT + r * 32 + c8);
    }
    cp_lds_wait();
    __syncthreads();
    bf16x16 af[2], bfrag[3];
#pragma unroll
    for (int i = 0; i < 2; ++i) af[i] = frag_a_lds(aT + (wm * 32 + i * 16) * 32, 32);
#pragma unroll
    for (int j = 0; j < 3; ++j) bfrag[j] = frag_b_lds(bT + (wn * 48 + j * 16) * 32, 32);
#pragma unroll
    for (int i = 0; i < 2; ++i)
#pragma unroll
      for (int j = 0; j < 3; ++j) acc[i][j] = wmma_bf16(af[i], bfrag[j], acc[i][j]);
    __syncthreads();
  }
  int lane = tid & 31, l15 = lane & 15, hi = lane >> 4;
  for (int i = 0; i < 2; ++i)
    for (int j = 0; j < 3; ++j) {
      int nl = wn * 48 + j * 16 + l15;
      float bv = bias[n0 + nl];
#pragma unroll
      for (int r = 0; r < 8; ++r) {
        int rl = wm * 32 + i * 16 + r + hi * 8;
        oT[rl * 96 + nl] = (bf16)(acc[i][j][r] + bv);
      }
    }
  __syncthreads();
  for (int i = tid; i < 128 * 12; i += 256) {   // coalesced b128 stores
    int r = i / 12, c8 = (i % 12) * 8;
    *(u32x4*)(qkv + (size_t)(m0 + r) * (3 * C_) + n0 + c8) =
        *(const u32x4*)(oT + r * 96 + c8);
  }
}

DEVINL int region3(int p, int lim) {  // 0/1/2 region label along one axis
  return (p < lim - WSZ) ? 0 : ((p < lim - SHIFT) ? 1 : 2);
}

// ---------------- K3: windowed attention, 1 block/window, 1 wave/head ------------
__global__ __launch_bounds__(128) void k_attention(const bf16* __restrict__ qkv,
                                                   const float* __restrict__ rpb,
                                                   bf16* __restrict__ ao) {
  __shared__ __align__(16) bf16 sQK[HEADS][64 * 32 * 2]; // 32KB q|k, reused as P; [0] reused as oT
  __shared__ __align__(16) bf16 sV[HEADS][64 * 32];      // 16KB v row-major (tok,d)
  __shared__ float sRPB[RPB_N];                          // 5.4KB bias table
  int tid = threadIdx.x;
  int h = tid >> 5, lane = tid & 31;
  int win = blockIdx.x;
  int rem = win & (NW - 1);
  int wd = rem >> 8, wh = (rem >> 4) & 15, ww = rem & 15;
  bf16* q = sQK[h];
  bf16* kk = q + 64 * 32;
  bf16* P = q;                        // reuse q|k region for probabilities
  bf16* vv = sV[h];
  for (int i = tid; i < RPB_N; i += 128) sRPB[i] = rpb[i];
  const bf16* base = qkv + (size_t)win * N_TOK * (3 * C_);
  u32x4 zv; zv.x = zv.y = zv.z = zv.w = 0u;
#pragma unroll
  for (int tt = 0; tt < 2; ++tt) {             // each lane loads 2 token rows
    int t = lane + tt * 32;
    const bf16* rp = base + (size_t)t * (3 * C_) + h * HD;  // 16B aligned
#pragma unroll
    for (int c8 = 0; c8 < 3; ++c8) {
      cp_b128_to_lds(rp + c8 * 8,          q + t * 32 + c8 * 8);
      cp_b128_to_lds(rp + C_ + c8 * 8,     kk + t * 32 + c8 * 8);
      cp_b128_to_lds(rp + 2 * C_ + c8 * 8, vv + t * 32 + c8 * 8);
    }
    *(u32x4*)(q + t * 32 + 24) = zv;           // pad hd 24 -> 32
    *(u32x4*)(kk + t * 32 + 24) = zv;
    *(u32x4*)(vv + t * 32 + 24) = zv;
  }
  cp_lds_wait();
  __syncthreads();
  // S = Q @ K^T  (hd padded to 32 -> one WMMA per 16x16 tile)
  bf16x16 qf[4], kf[4];
#pragma unroll
  for (int i = 0; i < 4; ++i) qf[i] = frag_a_lds(q + i * 16 * 32, 32);
#pragma unroll
  for (int j = 0; j < 4; ++j) kf[j] = frag_b_lds(kk + j * 16 * 32, 32);
  f32x8 s[4][4];
#pragma unroll
  for (int i = 0; i < 4; ++i)
#pragma unroll
    for (int j = 0; j < 4; ++j) s[i][j] = wmma_bf16(qf[i], kf[j], zero8());
  int l15 = lane & 15, hi = lane >> 4;
  // hoisted per-column (m) decomposition + region label
  int dmv[4], hmv[4], wmv[4], cmv[4];
#pragma unroll
  for (int j = 0; j < 4; ++j) {
    int mm = j * 16 + l15;
    int dm = mm >> 4, hm = (mm >> 2) & 3, wm = mm & 3;
    dmv[j] = dm; hmv[j] = hm; wmv[j] = wm;
    cmv[j] = region3(wd * 4 + dm, D_) * 9 + region3(wh * 4 + hm, H_) * 3 +
             region3(ww * 4 + wm, W_);
  }
  // scale + rel-pos bias (LDS) + shifted-window mask (analytic)
#pragma unroll
  for (int i = 0; i < 4; ++i)
#pragma unroll
    for (int r = 0; r < 8; ++r) {
      int n = i * 16 + r + hi * 8;
      int dn = n >> 4, hn = (n >> 2) & 3, wn = n & 3;
      int cn = region3(wd * 4 + dn, D_) * 9 + region3(wh * 4 + hn, H_) * 3 +
               region3(ww * 4 + wn, W_);
#pragma unroll
      for (int j = 0; j < 4; ++j) {
        int idx = (dn - dmv[j] + 3) * 49 + (hn - hmv[j] + 3) * 7 + (wn - wmv[j] + 3);
        float bias = sRPB[idx * HEADS + h];
        float msk = (cn != cmv[j]) ? -100.f : 0.f;
        s[i][j][r] = s[i][j][r] * SCALE + bias + msk;
      }
    }
  // row softmax: 64 cols over 4 tiles x 16 lanes (same half-wave)
  for (int i = 0; i < 4; ++i)
    for (int r = 0; r < 8; ++r) {
      float mx = fmaxf(fmaxf(s[i][0][r], s[i][1][r]), fmaxf(s[i][2][r], s[i][3][r]));
      for (int o = 8; o >= 1; o >>= 1) mx = fmaxf(mx, __shfl_xor(mx, o, 32));
      float sum = 0.f;
      for (int j = 0; j < 4; ++j) { float e = __expf(s[i][j][r] - mx); s[i][j][r] = e; sum += e; }
      for (int o = 8; o >= 1; o >>= 1) sum += __shfl_xor(sum, o, 32);
      float inv = 1.f / sum;
      for (int j = 0; j < 4; ++j) s[i][j][r] *= inv;
    }
  __syncthreads();
  // P -> LDS bf16 (row-major 64x64, overwrites q|k region)
  for (int i = 0; i < 4; ++i)
    for (int j = 0; j < 4; ++j)
      for (int r = 0; r < 8; ++r) {
        int n = i * 16 + r + hi * 8, mm = j * 16 + l15;
        P[n * 64 + mm] = (bf16)s[i][j][r];
      }
  __syncthreads();
  // O = P @ V  (K=64 -> 2 steps); V row-major (k,n) -> paired-halfword B frags
  f32x8 o[4][2];
  for (int i = 0; i < 4; ++i) for (int dt = 0; dt < 2; ++dt) o[i][dt] = zero8();
  for (int kt = 0; kt < 2; ++kt) {
    bf16x16 pf[4], vf[2];
#pragma unroll
    for (int i = 0; i < 4; ++i) pf[i] = frag_a_lds(P + i * 16 * 64 + kt * 32, 64);
#pragma unroll
    for (int dt = 0; dt < 2; ++dt) vf[dt] = frag_b_kxn(vv + (kt * 32) * 32 + dt * 16, 32);
#pragma unroll
    for (int i = 0; i < 4; ++i)
#pragma unroll
      for (int dt = 0; dt < 2; ++dt) o[i][dt] = wmma_bf16(pf[i], vf[dt], o[i][dt]);
  }
  __syncthreads();
  // transpose O through LDS then write coalesced
  bf16* oT = sQK[0];                 // 64*96 elems = 12KB, spills from sQK[0] into half of sQK[1]
  for (int i = 0; i < 4; ++i)
    for (int dt = 0; dt < 2; ++dt) {
      int d = dt * 16 + l15;
      if (d < HD)
        for (int r = 0; r < 8; ++r) {
          int n = i * 16 + r + hi * 8;
          oT[n * 96 + h * HD + d] = (bf16)o[i][dt][r];
        }
    }
  __syncthreads();
  for (int i = tid; i < 64 * 12; i += 128) {
    int r = i / 12, c8 = (i % 12) * 8;
    *(u32x4*)(ao + ((size_t)win * N_TOK + r) * C_ + c8) =
        *(const u32x4*)(oT + r * 96 + c8);
  }
}

// ---------------- K4: proj GEMM + window reverse + roll(+SHIFT) + residual -------
__global__ __launch_bounds__(256) void k_proj_residual(const bf16* __restrict__ ao,
                                                       const bf16* __restrict__ wp,
                                                       const float* __restrict__ pb,
                                                       const float* __restrict__ x,
                                                       float* __restrict__ y) {
  __shared__ __align__(16) bf16 aT[128 * 32];
  __shared__ __align__(16) bf16 bT[96 * 32];
  int tid = threadIdx.x;
  int wave = tid >> 5, wm = wave >> 1, wn = wave & 1;
  int m0 = blockIdx.x * 128;
  f32x8 acc[2][3];
  for (int i = 0; i < 2; ++i) for (int j = 0; j < 3; ++j) acc[i][j] = zero8();
  for (int ks = 0; ks < 3; ++ks) {
    for (int i = tid; i < 512; i += 256) {
      int r = i >> 2, c8 = (i & 3) * 8;
      cp_b128_to_lds(ao + (size_t)(m0 + r) * C_ + ks * 32 + c8, aT + r * 32 + c8);
    }
    for (int i = tid; i < 384; i += 256) {
      int r = i >> 2, c8 = (i & 3) * 8;
      cp_b128_to_lds(wp + (size_t)r * C_ + ks * 32 + c8, bT + r * 32 + c8);
    }
    cp_lds_wait();
    __syncthreads();
    bf16x16 af[2], bfrag[3];
#pragma unroll
    for (int i = 0; i < 2; ++i) af[i] = frag_a_lds(aT + (wm * 32 + i * 16) * 32, 32);
#pragma unroll
    for (int j = 0; j < 3; ++j) bfrag[j] = frag_b_lds(bT + (wn * 48 + j * 16) * 32, 32);
#pragma unroll
    for (int i = 0; i < 2; ++i)
#pragma unroll
      for (int j = 0; j < 3; ++j) acc[i][j] = wmma_bf16(af[i], bfrag[j], acc[i][j]);
    __syncthreads();
  }
  int lane = tid & 31, l15 = lane & 15, hi = lane >> 4;
  for (int i = 0; i < 2; ++i)
    for (int j = 0; j < 3; ++j) {
      int ng = wn * 48 + j * 16 + l15;   // channel 0..95
      float bv = pb[ng];
      for (int r = 0; r < 8; ++r) {
        int row = m0 + wm * 32 + i * 16 + r + hi * 8;
        int win = row >> 6, tok = row & 63;
        int b = win >> 11, rm = win & (NW - 1);
        int wd = rm >> 8, wh = (rm >> 4) & 15, ww = rm & 15;
        int dz = tok >> 4, hy = (tok >> 2) & 3, wx = tok & 3;
        int d = (wd * 4 + dz + SHIFT) & (D_ - 1);
        int hh = (wh * 4 + hy + SHIFT) & (H_ - 1);
        int wpp = (ww * 4 + wx + SHIFT) & (W_ - 1);
        int pos = (d << 12) | (hh << 6) | wpp;
        float sc = x[((size_t)b * C_ + ng) * DHW + pos];
        y[((size_t)b * DHW + pos) * C_ + ng] = sc + acc[i][j][r] + bv;
      }
    }
}

// ---------------- K5: fused LN2 + fc1 + GELU + fc2 + residual (channels-first out)
__global__ __launch_bounds__(256) void k_mlp(const float* __restrict__ y,
                                             const float* __restrict__ g,
                                             const float* __restrict__ bt,
                                             const bf16* __restrict__ w1,
                                             const float* __restrict__ b1,
                                             const bf16* __restrict__ w2,
                                             const float* __restrict__ b2,
                                             float* __restrict__ out) {
  __shared__ float yf[64 * C_];                  // 24KB residual rows
  __shared__ __align__(16) bf16 lnb[64 * C_];    // 12KB LN2(x) bf16
  __shared__ __align__(16) bf16 hT[64 * 32];     // 4KB hidden chunk
  __shared__ __align__(16) bf16 w1T[32 * C_];    // 6KB fc1 chunk (n,k)
  __shared__ __align__(16) bf16 w2T[C_ * 32];    // 6KB fc2 chunk (n,k)
  int tid = threadIdx.x;
  size_t t0 = (size_t)blockIdx.x * 64;
  for (int i = tid; i < 64 * C_ / 4; i += 256)
    cp_b128_to_lds(y + t0 * C_ + i * 4, yf + i * 4);
  cp_lds_wait();
  __syncthreads();
  if (tid < 64) {
    float s = 0.f, s2 = 0.f;
    for (int c = 0; c < C_; ++c) { float v = yf[tid * C_ + c]; s += v; s2 += v * v; }
    float m = s * (1.0f / C_), var = s2 * (1.0f / C_) - m * m;
    float rs = rsqrtf(var + 1e-5f);
    for (int c = 0; c < C_; ++c)
      lnb[tid * C_ + c] = (bf16)((yf[tid * C_ + c] - m) * rs * g[c] + bt[c]);
  }
  __syncthreads();
  int wave = tid >> 5, lane = tid & 31, l15 = lane & 15, hi = lane >> 4;
  int mt = wave >> 1, sub = wave & 1;
  f32x8 acc[3];
  for (int j = 0; j < 3; ++j) acc[j] = zero8();
  for (int kt = 0; kt < 12; ++kt) {          // stream 384 hidden in 32-chunks
    for (int i = tid; i < 384; i += 256) {   // w1 chunk 32x96
      int r = i / 12, c8 = (i % 12) * 8;
      cp_b128_to_lds(w1 + (size_t)(kt * 32 + r) * C_ + c8, w1T + r * C_ + c8);
    }
    for (int i = tid; i < 384; i += 256) {   // w2 chunk 96x32
      int r = i >> 2, c8 = (i & 3) * 8;
      cp_b128_to_lds(w2 + (size_t)r * (4 * C_) + kt * 32 + c8, w2T + r * 32 + c8);
    }
    cp_lds_wait();
    __syncthreads();
    f32x8 ha = zero8();
#pragma unroll
    for (int ks = 0; ks < 3; ++ks) {
      bf16x16 a = frag_a_lds(lnb + (mt * 16) * C_ + ks * 32, C_);
      bf16x16 bfr = frag_b_lds(w1T + (sub * 16) * C_ + ks * 32, C_);
      ha = wmma_bf16(a, bfr, ha);
    }
    {
      int n = sub * 16 + l15;
      float bv = b1[kt * 32 + n];
#pragma unroll
      for (int r = 0; r < 8; ++r) {
        float v = ha[r] + bv;
        float ge = 0.5f * v * (1.f + erff(v * 0.70710678f));   // exact GELU
        hT[(mt * 16 + r + hi * 8) * 32 + n] = (bf16)ge;
      }
    }
    __syncthreads();
    bf16x16 a2 = frag_a_lds(hT + (mt * 16) * 32, 32);
#pragma unroll
    for (int j = 0; j < 3; ++j) {
      bf16x16 b2f = frag_b_lds(w2T + ((sub * 3 + j) * 16) * 32, 32);
      acc[j] = wmma_bf16(a2, b2f, acc[j]);
    }
    __syncthreads();
  }
  int b = (int)(t0 >> 17);
  int pos0 = (int)(t0 & (DHW - 1));
  for (int j = 0; j < 3; ++j) {
    int n = (sub * 3 + j) * 16 + l15;
    float bv = b2[n];
    float ov[8];
#pragma unroll
    for (int r = 0; r < 8; ++r) {
      int row = mt * 16 + r + hi * 8;
      ov[r] = acc[j][r] + bv + yf[row * C_ + n];
    }
    float* dst = out + ((size_t)b * C_ + n) * DHW + pos0 + mt * 16 + hi * 8;
    *(f32x4*)(dst)     = *(const f32x4*)(ov);      // 8 consecutive rows: 2x b128
    *(f32x4*)(dst + 4) = *(const f32x4*)(ov + 4);
  }
}

// ---------------- launcher ----------------
extern "C" void kernel_launch(void* const* d_in, const int* in_sizes, int n_in,
                              void* d_out, int out_size, void* d_ws, size_t ws_size,
                              hipStream_t stream) {
  (void)in_sizes; (void)n_in; (void)out_size; (void)ws_size;
  const float* x     = (const float*)d_in[0];
  const float* n1g   = (const float*)d_in[1];
  const float* n1b   = (const float*)d_in[2];
  const float* qkvw  = (const float*)d_in[3];
  const float* qkvb  = (const float*)d_in[4];
  const float* rpb   = (const float*)d_in[5];
  const float* projw = (const float*)d_in[6];
  const float* projb = (const float*)d_in[7];
  const float* n2g   = (const float*)d_in[8];
  const float* n2b   = (const float*)d_in[9];
  const float* fc1w  = (const float*)d_in[10];
  const float* fc1b  = (const float*)d_in[11];
  const float* fc2w  = (const float*)d_in[12];
  const float* fc2b  = (const float*)d_in[13];

  char* ws = (char*)d_ws;
  bf16* w_qkv  = (bf16*)(ws + OFF_QKVW);
  bf16* w_proj = (bf16*)(ws + OFF_PROJW);
  bf16* w_fc1  = (bf16*)(ws + OFF_FC1W);
  bf16* w_fc2  = (bf16*)(ws + OFF_FC2W);
  bf16* xw     = (bf16*)(ws + OFF_XW);
  bf16* qkv    = (bf16*)(ws + OFF_QKV);
  bf16* ao     = (bf16*)(ws + OFF_AO);
  float* yv    = (float*)(ws + OFF_Y);
  float* out   = (float*)d_out;

  k_convert_weights<<<144, 256, 0, stream>>>(qkvw, projw, fc1w, fc2w,
                                             w_qkv, w_proj, w_fc1, w_fc2);
  k_ln1_window<<<2048, 128, 0, stream>>>(x, n1g, n1b, xw);
  k_qkv_gemm<<<dim3(ROWS / 128, 3), 256, 0, stream>>>(xw, w_qkv, qkvb, qkv);
  k_attention<<<B_ * NW, 128, 0, stream>>>(qkv, rpb, ao);
  k_proj_residual<<<ROWS / 128, 256, 0, stream>>>(ao, w_proj, projb, x, yv);
  k_mlp<<<ROWS / 64, 256, 0, stream>>>(yv, n2g, n2b, w_fc1, fc1b, w_fc2, fc2b, out);
}
